// Head_4587025072851
// MI455X (gfx1250) — matrix operand order repaired and checked by reading, not compiled, not run on hardware
//
#include <hip/hip_runtime.h>
#include <hip/hip_bf16.h>

typedef _Float16 v16h __attribute__((ext_vector_type(16)));
typedef _Float16 v8h  __attribute__((ext_vector_type(8)));
typedef float    v8f  __attribute__((ext_vector_type(8)));
typedef float    v4f  __attribute__((ext_vector_type(4)));
typedef unsigned int v4u __attribute__((ext_vector_type(4)));

#define BS 4
#define SL 4096
#define ES 1024
#define HS 128
#define ROWS (BS * SL)          // 16384 total sequence rows

static __device__ __forceinline__ v8f wmma_f16(v16h a, v16h b, v8f c) {
  // D = A(16x32 f16) * B(32x16 f16) + C(16x16 f32)
  return __builtin_amdgcn_wmma_f32_16x16x32_f16(
      /*neg_a=*/false, a, /*neg_b=*/false, b,
      /*c_mod=*/(short)0, c, /*reuse_a=*/false, /*reuse_b=*/false);
}

union Frag16 { v16h h; v4u u[2]; };
union Pack8  { v8h  h; v4u u;    };

// A-operand fragment (16-bit, 16x32): lane holds row M=lane%16,
// K = {8*half .. 8*half+7} in VGPR0-3 and {16+8*half ..} in VGPR4-7.
// p points at row start + 8*half elements -> two 16B chunks at +0 and +16.
static __device__ __forceinline__ v16h load_frag_a(const _Float16* p) {
  Frag16 f;
  f.u[0] = *(const v4u*)(p);
  f.u[1] = *(const v4u*)(p + 16);
  return f.h;
}

// B-operand fragment (16-bit, 32x16): lane holds col N=lane%16,
// K = {8*half .. 8*half+7} and {16+8*half ..}; p points at col start + 8*half.
static __device__ __forceinline__ v16h load_frag_b(const _Float16* p) {
  Frag16 f;
  f.u[0] = *(const v4u*)(p);
  f.u[1] = *(const v4u*)(p + 16);
  return f.h;
}

// A-operand fragment converted from fp32 source (same layout as load_frag_a).
static __device__ __forceinline__ v16h cvt_frag_a(const float* p) {
  v4f a0 = *(const v4f*)(p);
  v4f a1 = *(const v4f*)(p + 4);
  v4f b0 = *(const v4f*)(p + 16);
  v4f b1 = *(const v4f*)(p + 20);
  v16h h;
#pragma unroll
  for (int i = 0; i < 4; ++i) {
    h[i]      = (_Float16)a0[i];
    h[4 + i]  = (_Float16)a1[i];
    h[8 + i]  = (_Float16)b0[i];
    h[12 + i] = (_Float16)b1[i];
  }
  return h;
}

static __device__ __forceinline__ v8f vzero8() {
  v8f z = {0.f, 0.f, 0.f, 0.f, 0.f, 0.f, 0.f, 0.f};
  return z;
}

// ---------------------------------------------------------------------------
// Kernel 0: transpose + fp32->f16 convert the weights: Wt[mat][n][k] = W[k][n]
// Fold the attention scale H^-0.5 into Wq.
// ---------------------------------------------------------------------------
__global__ __launch_bounds__(256) void wt_prep(const float* __restrict__ Wq,
                                               const float* __restrict__ Wk,
                                               const float* __restrict__ Wv,
                                               _Float16* __restrict__ Wt) {
  int idx = blockIdx.x * 256 + threadIdx.x;      // 0 .. 3*ES*HS-1
  int mat = idx >> 17;                           // / (ES*HS) = /131072
  int i   = idx & (ES * HS - 1);
  int k   = i >> 7;                              // / HS
  int n   = i & (HS - 1);
  const float* W = (mat == 0) ? Wq : (mat == 1) ? Wk : Wv;
  float s = (mat == 0) ? 0.08838834764831845f : 1.0f;   // 128^-0.5 folded into Wq
  Wt[(size_t)mat * (ES * HS) + (size_t)n * ES + k] = (_Float16)(W[i] * s);
}

// ---------------------------------------------------------------------------
// Kernel 1: QKV projection. One wave computes one 16x128 output row-tile of
// one of {Q,K,V}. Software-pipelined: B fragments issued as a batch, the
// next k-step's A fragment (fp32 load + cvt) overlaps the WMMA chain.
// Q,K stored row-major f16 [ROWS][HS]; V stored transposed f16 [BS][HS][SL].
// ---------------------------------------------------------------------------
__global__ __launch_bounds__(128) void qkv_proj(const float* __restrict__ x,
                                                const _Float16* __restrict__ Wt,
                                                _Float16* __restrict__ Q,
                                                _Float16* __restrict__ K,
                                                _Float16* __restrict__ Vt) {
  const int lane = threadIdx.x & 31;
  const int wave = threadIdx.x >> 5;
  const int task = blockIdx.x * 4 + wave;        // 0 .. 3071
  const int mat  = task >> 10;                   // 0=Q,1=K,2=V
  const int rt   = task & 1023;                  // row-tile index
  const int m0   = rt * 16;                      // first global row
  const int col  = lane & 15;
  const int half = lane >> 4;

  const _Float16* W = Wt + (size_t)mat * (ES * HS) +
                      (size_t)col * ES + 8 * half;

  v8f acc[8];
#pragma unroll
  for (int t = 0; t < 8; ++t) acc[t] = vzero8();

  const float* arow = x + (size_t)(m0 + col) * ES + 8 * half;

  v16h afrag = cvt_frag_a(arow);                 // prologue: k0 = 0

  for (int k0 = 0; k0 < ES; k0 += 32) {
    // batch-issue all 8 B fragments (one load clause, staggered waits)
    v16h bf[8];
#pragma unroll
    for (int t = 0; t < 8; ++t) bf[t] = load_frag_b(W + (size_t)(t * 16) * ES + k0);

    // issue next A fragment loads before consuming this k-step
    v16h anext;
    if (k0 + 32 < ES) anext = cvt_frag_a(arow + k0 + 32);

#pragma unroll
    for (int t = 0; t < 8; ++t) acc[t] = wmma_f16(afrag, bf[t], acc[t]);
    afrag = anext;
  }

  if (mat < 2) {
    // row-major [ROWS][HS]; C/D layout: lane=col n, VGPR r = row r+8*half
    _Float16* dst = (mat == 0) ? Q : K;
#pragma unroll
    for (int t = 0; t < 8; ++t)
#pragma unroll
      for (int r = 0; r < 8; ++r)
        dst[(size_t)(m0 + r + 8 * half) * HS + t * 16 + col] = (_Float16)acc[t][r];
  } else {
    // transposed V: Vt[b][h][s]; 8 consecutive s per lane -> one 16B store
    int b  = m0 >> 12;            // / SL
    int s0 = m0 & (SL - 1);
#pragma unroll
    for (int t = 0; t < 8; ++t) {
      Pack8 p;
#pragma unroll
      for (int r = 0; r < 8; ++r) p.h[r] = (_Float16)acc[t][r];
      _Float16* dst = Vt + ((size_t)(b * HS + t * 16 + col) * SL + s0 + 8 * half);
      *(v4u*)dst = p.u;
    }
  }
}

// ---------------------------------------------------------------------------
// Kernel 2: causal flash attention. One wave per 16-row query tile, key
// blocks of 32, online softmax, WMMA for QK^T and PV. K fragments batched
// ahead of the QK WMMA chain; V fragments issued before the softmax VALU
// work so their latency is hidden.
// ---------------------------------------------------------------------------
__global__ __launch_bounds__(128) void attn_fa(const _Float16* __restrict__ Q,
                                               const _Float16* __restrict__ K,
                                               const _Float16* __restrict__ Vt,
                                               float* __restrict__ out) {
  __shared__ _Float16 lds_p[4][16 * 32];         // wave-private P transpose buffer

  const int lane = threadIdx.x & 31;
  const int wave = threadIdx.x >> 5;
  const int qt   = blockIdx.x * 4 + wave;        // 0 .. 1023
  const int b    = qt >> 8;                      // / (SL/16)
  const int s0   = (qt & 255) * 16;              // first query row in batch
  const int col  = lane & 15;
  const int half = lane >> 4;

  const _Float16* Kb = K  + (size_t)b * SL * HS + (size_t)col * HS + 8 * half;
  const _Float16* Vb = Vt + (size_t)b * HS * SL + (size_t)col * SL + 8 * half;
  _Float16*      pbuf = lds_p[wave];

  // Q tile (16x128) as 4 A-fragments, loaded once
  const size_t qrow = (size_t)(b * SL + s0 + col) * HS;
  v16h qf[4];
#pragma unroll
  for (int f = 0; f < 4; ++f) qf[f] = load_frag_a(Q + qrow + f * 32 + 8 * half);

  v8f o[8];
#pragma unroll
  for (int t = 0; t < 8; ++t) o[t] = vzero8();
  float mrow[8], lrow[8];
#pragma unroll
  for (int r = 0; r < 8; ++r) { mrow[r] = -3.0e38f; lrow[r] = 0.0f; }

  const float c_log2e = 1.4426950408889634f;
  const int nblk = (s0 >> 5) + 1;                // causal: keys 0 .. s0+15

  for (int blk = 0; blk < nblk; ++blk) {
    const int n0 = blk * 32;
    if (blk + 1 < nblk)                          // prefetch next K block
      __builtin_prefetch(Kb + (size_t)(n0 + 32) * HS, 0, 1);

    // ---- batch-issue all 8 K fragments, then QK^T WMMA chain ----
    v16h kb[8];
#pragma unroll
    for (int f = 0; f < 4; ++f)
#pragma unroll
      for (int t = 0; t < 2; ++t)
        kb[f * 2 + t] = load_frag_b(Kb + (size_t)(n0 + t * 16) * HS + f * 32);

    v8f s[2];
    s[0] = vzero8(); s[1] = vzero8();
#pragma unroll
    for (int f = 0; f < 4; ++f) {
      s[0] = wmma_f16(qf[f], kb[f * 2 + 0], s[0]);
      s[1] = wmma_f16(qf[f], kb[f * 2 + 1], s[1]);
    }

    // ---- issue V fragments now; softmax VALU hides their latency ----
    v16h vb[8];
#pragma unroll
    for (int t = 0; t < 8; ++t)
      vb[t] = load_frag_b(Vb + (size_t)(t * 16) * SL + n0);

    // ---- causal mask (diagonal block only; wave-uniform branch) ----
    if (blk == nblk - 1) {
#pragma unroll
      for (int t = 0; t < 2; ++t)
#pragma unroll
        for (int r = 0; r < 8; ++r) {
          int c  = n0 + t * 16 + col;
          int rg = s0 + r + 8 * half;
          if (c > rg) s[t][r] = -3.0e38f;
        }
    }

    // ---- online softmax update ----
    float alpha[8];
#pragma unroll
    for (int r = 0; r < 8; ++r) {
      float v = fmaxf(s[0][r], s[1][r]);
      v = fmaxf(v, __shfl_xor(v, 1, 32));
      v = fmaxf(v, __shfl_xor(v, 2, 32));
      v = fmaxf(v, __shfl_xor(v, 4, 32));
      v = fmaxf(v, __shfl_xor(v, 8, 32));       // row max within 16-lane group
      float mnew = fmaxf(mrow[r], v);
      alpha[r]   = exp2f((mrow[r] - mnew) * c_log2e);
      float p0   = exp2f((s[0][r] - mnew) * c_log2e);
      float p1   = exp2f((s[1][r] - mnew) * c_log2e);
      s[0][r] = p0; s[1][r] = p1;
      mrow[r] = mnew;
      float ps = p0 + p1;
      ps += __shfl_xor(ps, 1, 32);
      ps += __shfl_xor(ps, 2, 32);
      ps += __shfl_xor(ps, 4, 32);
      ps += __shfl_xor(ps, 8, 32);
      lrow[r] = lrow[r] * alpha[r] + ps;
    }
#pragma unroll
    for (int t = 0; t < 8; ++t)
#pragma unroll
      for (int r = 0; r < 8; ++r) o[t][r] *= alpha[r];

    // ---- transpose P (D-layout -> A-layout) through wave-private LDS ----
#pragma unroll
    for (int t = 0; t < 2; ++t)
#pragma unroll
      for (int r = 0; r < 8; ++r)
        pbuf[(r + 8 * half) * 32 + t * 16 + col] = (_Float16)s[t][r];
    asm volatile("s_wait_dscnt 0" ::: "memory"); // in-wave DS RAW ordering
    v16h pf = load_frag_a(pbuf + col * 32 + 8 * half);

    // ---- O += P * V ----
#pragma unroll
    for (int t = 0; t < 8; ++t) o[t] = wmma_f16(pf, vb[t], o[t]);
  }

  // ---- normalize + write fp32 output [BS][SL][HS] ----
  float invl[8];
#pragma unroll
  for (int r = 0; r < 8; ++r) invl[r] = 1.0f / lrow[r];
#pragma unroll
  for (int t = 0; t < 8; ++t)
#pragma unroll
    for (int r = 0; r < 8; ++r)
      out[(size_t)(b * SL + s0 + r + 8 * half) * HS + t * 16 + col] =
          o[t][r] * invl[r];
}

// ---------------------------------------------------------------------------
extern "C" void kernel_launch(void* const* d_in, const int* in_sizes, int n_in,
                              void* d_out, int out_size, void* d_ws, size_t ws_size,
                              hipStream_t stream) {
  const float* x  = (const float*)d_in[0];
  const float* Wq = (const float*)d_in[1];
  const float* Wk = (const float*)d_in[2];
  const float* Wv = (const float*)d_in[3];
  float* out = (float*)d_out;

  char* ws = (char*)d_ws;
  _Float16* Wt =  (_Float16*)(ws);                              // 3*128*1024 f16 (768 KB)
  _Float16* Qh =  (_Float16*)(ws + (1u << 20));                 // 16384*128 f16 (4 MB)
  _Float16* Kh =  (_Float16*)(ws + (5u << 20));                 // 4 MB
  _Float16* Vt =  (_Float16*)(ws + (9u << 20));                 // [BS][HS][SL] f16 (4 MB)

  wt_prep <<<(3 * ES * HS) / 256, 256, 0, stream>>>(Wq, Wk, Wv, Wt);
  qkv_proj<<<(3 * (ROWS / 16)) / 4, 128, 0, stream>>>(x, Wt, Qh, Kh, Vt);
  attn_fa <<<(ROWS / 16) / 4, 128, 0, stream>>>(Qh, Kh, Vt, out);
}
